// TransformerNLI_22746146800365
// MI455X (gfx1250) — matrix operand order repaired and checked by reading, not compiled, not run on hardware
//
#include <hip/hip_runtime.h>
#include <hip/hip_bf16.h>
#include <stdint.h>

#define Bz 64
#define Ssz 256
#define Ez 512
#define Hh 8
#define FFNsz 2048
#define NLAY 3
#define PD 300
#define UNITSZ 3
#define HEADsz 64

typedef _Float16 h16;
typedef __attribute__((ext_vector_type(16))) _Float16 v16h;
typedef __attribute__((ext_vector_type(8)))  float    v8f;
typedef int v4i __attribute__((vector_size(4 * sizeof(int))));

#define AS1 __attribute__((address_space(1)))
#define AS3 __attribute__((address_space(3)))

// gfx1250 async memory->LDS staging, if the toolchain declares the builtins.
#if defined(__has_builtin)
#if __has_builtin(__builtin_amdgcn_global_load_async_to_lds_b128) && \
    __has_builtin(__builtin_amdgcn_s_wait_asynccnt)
#define USE_ASYNC_LDS 1
#endif
#endif
#ifndef USE_ASYNC_LDS
#define USE_ASYNC_LDS 0
#endif

union FragH { v16h v; unsigned int u[8]; };

// Fragment loaders (contiguous k order used by this code, same for A and B):
//   A 16x32: lane m = lrow; lanes 0-15 K=0..15, lanes 16-31 K=16..31
//   B 32x16: lane n = lrow; lanes 0-15 K=0..15, lanes 16-31 K=16..31
__device__ __forceinline__ void load_fragA(FragH& f, const h16* rowp, int grp) {
#pragma unroll
  for (int i = 0; i < 8; ++i) {
    const int ka = 2 * i + grp * 16;
    f.u[i] = *(const unsigned int*)(rowp + ka);
  }
}
__device__ __forceinline__ void load_fragB(FragH& f, const h16* rowp, int grp) {
#pragma unroll
  for (int i = 0; i < 8; ++i) {
    const int kb = 2 * i + grp * 16;
    f.u[i] = *(const unsigned int*)(rowp + kb);
  }
}

// ---------------------------------------------------------------------------
// fp32 -> f16 tensor conversion (vectorized x4)
// ---------------------------------------------------------------------------
__global__ __launch_bounds__(256) void cvt32to16_kernel(const float* __restrict__ in,
                                                        h16* __restrict__ out, int64_t n)
{
  int64_t i = ((int64_t)blockIdx.x * 256 + threadIdx.x) * 4;
  if (i + 4 <= n) {
    const float4 f = *(const float4*)(in + i);
    __align__(8) h16 h[4] = {(h16)f.x, (h16)f.y, (h16)f.z, (h16)f.w};
    *(uint2*)(out + i) = *(const uint2*)h;
  } else {
    for (; i < n; ++i) out[i] = (h16)in[i];
  }
}

// ---------------------------------------------------------------------------
// FAST f16 WMMA GEMM (no bounds checks): requires M%128==0, N%64==0, K%32==0.
//   C[M,N] = scale*(A[M,K] @ Bw^T + bias) (relu?) (+resid? fp32 only)
//   A, Bw f16; Bw is [N,K] row-major (weight form).
//   OUT16=1: C is f16. OUT16=0: C is fp32 (+ optional fp32 residual add).
//   256 threads = 8 waves; macro-tile 128x64; each wave owns a 32x32 sub-tile
//   => 4 accumulators, 4 v_wmma_f32_16x16x32_f16 per K-step.
//   Staging uses GLOBAL_LOAD_ASYNC_TO_LDS_B128 (ASYNCcnt) when available.
// ---------------------------------------------------------------------------
template <int OUT16>
__global__ __launch_bounds__(256) void gemm_fast16_kernel(
    const h16* __restrict__ A, int64_t lda, int64_t aBS,
    const h16* __restrict__ Bw, int64_t ldb, int64_t bBS,
    const float* __restrict__ bias,
    const float* resid,
    void* Cp, int64_t ldc, int64_t cBS,
    int K, float scale, int relu)
{
  __shared__ __align__(16) h16 At[128 * 32];  // 8 KB
  __shared__ __align__(16) h16 Bt[64 * 32];   // 4 KB

  const int tid  = threadIdx.x;
  const int wave = tid >> 5;
  const int lane = tid & 31;
  const int lrow = lane & 15;
  const int grp  = lane >> 4;
  const int wm   = (wave & 3) * 32;
  const int wn   = (wave >> 2) * 32;

  const int64_t z = blockIdx.z;
  const int row0blk = blockIdx.y * 128;
  const int col0blk = blockIdx.x * 64;

  // staging source pointers
  const int ar  = tid >> 1;             // 0..127
  const int ac0 = (tid & 1) * 16;       // 0 / 16
  const h16* aSrc = A + z * aBS + (int64_t)(row0blk + ar) * lda + ac0;
  const int bn  = tid >> 2;             // 0..63
  const int bc0 = (tid & 3) * 8;        // 0,8,16,24
  const h16* bSrc = Bw + z * bBS + (int64_t)(col0blk + bn) * ldb + bc0;

  v8f acc00 = {}, acc01 = {}, acc10 = {}, acc11 = {};

  for (int k0 = 0; k0 < K; k0 += 32) {
#if USE_ASYNC_LDS
    // ---- async DMA: global -> LDS, 3x b128 per thread, tracked by ASYNCcnt ----
    __builtin_amdgcn_global_load_async_to_lds_b128(
        (AS1 v4i*)(aSrc + k0),     (AS3 v4i*)&At[ar * 32 + ac0],     0, 0);
    __builtin_amdgcn_global_load_async_to_lds_b128(
        (AS1 v4i*)(aSrc + k0 + 8), (AS3 v4i*)&At[ar * 32 + ac0 + 8], 0, 0);
    __builtin_amdgcn_global_load_async_to_lds_b128(
        (AS1 v4i*)(bSrc + k0),     (AS3 v4i*)&Bt[bn * 32 + bc0],     0, 0);
    __builtin_amdgcn_s_wait_asynccnt(0);
#else
    // ---- sync staging: b128 loads -> b128 LDS stores ----
    *(uint4*)&At[ar * 32 + ac0]     = *(const uint4*)(aSrc + k0);
    *(uint4*)&At[ar * 32 + ac0 + 8] = *(const uint4*)(aSrc + k0 + 8);
    __builtin_prefetch(aSrc + k0 + 32, 0, 1);
    *(uint4*)&Bt[bn * 32 + bc0] = *(const uint4*)(bSrc + k0);
    __builtin_prefetch(bSrc + k0 + 32, 0, 1);
#endif
    __syncthreads();

    FragH a0, a1, b0, b1;
    load_fragA(a0, &At[(wm + lrow) * 32], grp);
    load_fragA(a1, &At[(wm + 16 + lrow) * 32], grp);
    load_fragB(b0, &Bt[(wn + lrow) * 32], grp);
    load_fragB(b1, &Bt[(wn + 16 + lrow) * 32], grp);

    acc00 = __builtin_amdgcn_wmma_f32_16x16x32_f16(false, a0.v, false, b0.v, (short)0, acc00, false, false);
    acc01 = __builtin_amdgcn_wmma_f32_16x16x32_f16(false, a0.v, false, b1.v, (short)0, acc01, false, false);
    acc10 = __builtin_amdgcn_wmma_f32_16x16x32_f16(false, a1.v, false, b0.v, (short)0, acc10, false, false);
    acc11 = __builtin_amdgcn_wmma_f32_16x16x32_f16(false, a1.v, false, b1.v, (short)0, acc11, false, false);
    __syncthreads();
  }

  // ---- epilogue ----
  const int col0 = col0blk + wn + lrow;
  const int col1 = col0blk + wn + 16 + lrow;
  const int r0   = row0blk + wm + 8 * grp;
  const int r1   = row0blk + wm + 16 + 8 * grp;
  const float bv0 = bias ? bias[col0] : 0.f;
  const float bv1 = bias ? bias[col1] : 0.f;

  v8f accs[4] = {acc00, acc01, acc10, acc11};
  const int   rows[4] = {r0, r0, r1, r1};
  const int   cols[4] = {col0, col1, col0, col1};
  const float bvs[4]  = {bv0, bv1, bv0, bv1};
#pragma unroll
  for (int t = 0; t < 4; ++t) {
#pragma unroll
    for (int i = 0; i < 8; ++i) {
      float y = (accs[t][i] + bvs[t]) * scale;
      if (relu) y = fmaxf(y, 0.f);
      const int64_t idx = (int64_t)(rows[t] + i) * ldc + cols[t];
      if (OUT16) {
        ((h16*)Cp + z * cBS)[idx] = (h16)y;
      } else {
        if (resid) y += (resid + z * cBS)[idx];
        ((float*)Cp + z * cBS)[idx] = y;
      }
    }
  }
}

// ---------------------------------------------------------------------------
// GUARDED f16 WMMA GEMM (tail path: compare head M=64, logits N=3).
// Branch-free staging: clamp OOB indices to 0, load unconditionally, then
// select-zero -> no exec-mask divergence in the K loop.
// ---------------------------------------------------------------------------
__global__ __launch_bounds__(256) void gemm_guarded16_kernel(
    const h16* __restrict__ A, int64_t lda,
    const h16* __restrict__ Bw, int64_t ldb,
    const float* __restrict__ bias,
    void* Cp, int64_t ldc,
    int M, int N, int K, float scale, int relu, int out16)
{
  __shared__ __align__(16) h16 Bt[16 * 32];
  __shared__ __align__(16) h16 At[8][16 * 32];

  const int tid  = threadIdx.x;
  const int wave = tid >> 5;
  const int lane = tid & 31;
  const int lrow = lane & 15;
  const int grp  = lane >> 4;

  const int colBase = blockIdx.x * 16;
  const int rowBase = blockIdx.y * 128 + wave * 16;

  // precomputed clamped staging coordinates (uniform across K loop)
  const int sn  = tid >> 4;                    // 0..15 (B row within tile)
  const int skk = (tid & 15) * 2;              // 0..30
  const int gn  = colBase + sn;
  const int gns = (gn < N) ? gn : 0;
  const bool nOK = (gn < N);

  const int sr  = lane >> 1;                   // 0..15 (A row within tile)
  const int sc0 = (lane & 1) * 16;             // 0/16
  const int64_t grow = rowBase + sr;
  const int64_t grs  = (grow < (int64_t)M) ? grow : 0;
  const bool mOK = (grow < (int64_t)M);

  v8f acc = {};
  for (int k0 = 0; k0 < K; k0 += 32) {
    { // B tile 16x32
      h16 b0 = Bw[(int64_t)gns * ldb + (k0 + skk)];
      h16 b1 = Bw[(int64_t)gns * ldb + (k0 + skk + 1)];
      if (!nOK) { b0 = (h16)0; b1 = (h16)0; }
      Bt[sn * 32 + skk]     = b0;
      Bt[sn * 32 + skk + 1] = b1;
    }
    { // A tile 16x32 per wave
      const h16* ap = A + grs * lda + k0 + sc0;
      uint4 u0 = *(const uint4*)(ap);
      uint4 u1 = *(const uint4*)(ap + 8);
      if (!mOK) { u0.x = u0.y = u0.z = u0.w = 0u; u1.x = u1.y = u1.z = u1.w = 0u; }
      *(uint4*)&At[wave][sr * 32 + sc0]     = u0;
      *(uint4*)&At[wave][sr * 32 + sc0 + 8] = u1;
    }
    __syncthreads();

    FragH fa, fb;
    load_fragA(fa, &At[wave][lrow * 32], grp);
    load_fragB(fb, &Bt[lrow * 32], grp);
    acc = __builtin_amdgcn_wmma_f32_16x16x32_f16(
        false, fa.v, false, fb.v, (short)0, acc, false, false);
    __syncthreads();
  }

  const int col = colBase + lrow;
  const float bv = (bias && col < N) ? bias[col] : 0.f;
#pragma unroll
  for (int i = 0; i < 8; ++i) {
    const int row = rowBase + i + 8 * grp;
    if (row < M && col < N) {
      float y = (acc[i] + bv) * scale;
      if (relu) y = fmaxf(y, 0.f);
      const int64_t idx = (int64_t)row * ldc + col;
      if (out16) ((h16*)Cp)[idx] = (h16)y;
      else       ((float*)Cp)[idx] = y;
    }
  }
}

// ---------------------------------------------------------------------------
// LayerNorm over E=512, one token per block; fp32 in, f16 out.
// ---------------------------------------------------------------------------
__global__ __launch_bounds__(256) void ln512_kernel(const float* __restrict__ x,
                                                    const float* __restrict__ g,
                                                    const float* __restrict__ b,
                                                    h16* __restrict__ y)
{
  __shared__ float red[256];
  const int t = threadIdx.x;
  const int64_t base = (int64_t)blockIdx.x * Ez;
  float a0 = x[base + t];
  float a1 = x[base + t + 256];
  red[t] = a0 + a1;
  __syncthreads();
  for (int off = 128; off > 0; off >>= 1) {
    if (t < off) red[t] += red[t + off];
    __syncthreads();
  }
  const float mean = red[0] * (1.0f / Ez);
  __syncthreads();
  const float d0 = a0 - mean, d1 = a1 - mean;
  red[t] = d0 * d0 + d1 * d1;
  __syncthreads();
  for (int off = 128; off > 0; off >>= 1) {
    if (t < off) red[t] += red[t + off];
    __syncthreads();
  }
  const float rstd = rsqrtf(red[0] * (1.0f / Ez) + 1e-5f);
  y[base + t]       = (h16)(d0 * rstd * g[t] + b[t]);
  y[base + t + 256] = (h16)(d1 * rstd * g[t + 256] + b[t + 256]);
}

// ---------------------------------------------------------------------------
// Row softmax over S=256; fp32 scores in, f16 probabilities out.
// ---------------------------------------------------------------------------
__global__ __launch_bounds__(256) void softmax256_kernel(const float* __restrict__ x,
                                                         h16* __restrict__ p)
{
  __shared__ float red[256];
  const int t = threadIdx.x;
  const int64_t base = (int64_t)blockIdx.x * Ssz;
  const float v = x[base + t];
  red[t] = v;
  __syncthreads();
  for (int off = 128; off > 0; off >>= 1) {
    if (t < off) red[t] = fmaxf(red[t], red[t + off]);
    __syncthreads();
  }
  const float mx = red[0];
  __syncthreads();
  const float e = __expf(v - mx);
  red[t] = e;
  __syncthreads();
  for (int off = 128; off > 0; off >>= 1) {
    if (t < off) red[t] += red[t + off];
    __syncthreads();
  }
  p[base + t] = (h16)(e / red[0]);
}

// ---------------------------------------------------------------------------
// Per-head V transpose: vt[z][n][k] = v[k*(B*E) + z*64 + n]  (f16 -> f16)
// ---------------------------------------------------------------------------
__global__ void transposeV_kernel(const h16* __restrict__ v, h16* __restrict__ vt)
{
  const int64_t TOT = (int64_t)Bz * Hh * HEADsz * Ssz;
  const int64_t i = (int64_t)blockIdx.x * 256 + threadIdx.x;
  if (i >= TOT) return;
  const int n = (int)(i & (HEADsz - 1));
  const int k = (int)((i >> 6) & (Ssz - 1));
  const int z = (int)(i >> 14);
  vt[(int64_t)z * (HEADsz * Ssz) + (int64_t)n * Ssz + k] =
      v[(int64_t)k * (Bz * Ez) + z * HEADsz + n];
}

// ---------------------------------------------------------------------------
// Embedding gather + 300->512 projection + sinusoidal PE, out fp32 [S,B,E].
// ---------------------------------------------------------------------------
__global__ __launch_bounds__(256) void embed_kernel(const int* __restrict__ tok,
                                                    const float* __restrict__ emb,
                                                    const float* __restrict__ pw,
                                                    const float* __restrict__ pb,
                                                    float* __restrict__ out)
{
  __shared__ float er[PD];
  const int bs = blockIdx.x;        // b*S + s
  const int b = bs / Ssz;
  const int s = bs % Ssz;
  const int token = tok[bs];
  for (int i = threadIdx.x; i < PD; i += 256)
    er[i] = emb[(int64_t)token * PD + i];
  __syncthreads();
  const float pos = (float)s;
  for (int o = threadIdx.x; o < Ez; o += 256) {
    const float* wr = pw + (int64_t)o * PD;
    float acc = pb[o];
    for (int i = 0; i < PD; ++i) acc = fmaf(wr[i], er[i], acc);
    const float div = __expf(-(float)(o & ~1) * (9.21034037198f / (float)Ez));
    const float pe = (o & 1) ? __cosf(pos * div) : __sinf(pos * div);
    out[(int64_t)s * (Bz * Ez) + (int64_t)b * Ez + o] = acc + pe;
  }
}

// ---------------------------------------------------------------------------
// Concat along last dim (fp32 in -> f16 out), rows of width E each side.
// ---------------------------------------------------------------------------
__global__ void concat2_kernel(const float* __restrict__ a, const float* __restrict__ c,
                               h16* __restrict__ out, int64_t T)
{
  const int64_t i = (int64_t)blockIdx.x * 256 + threadIdx.x;
  const int64_t total = T * (int64_t)(2 * Ez);
  if (i >= total) return;
  const int64_t t = i / (2 * Ez);
  const int j = (int)(i % (2 * Ez));
  out[i] = (h16)((j < Ez) ? a[t * Ez + j] : c[t * Ez + (j - Ez)]);
}

// ---------------------------------------------------------------------------
// out[b,e] = (sum_s x[s,b,e]) * scale    (fp32 -> fp32)
// ---------------------------------------------------------------------------
__global__ void sumscale_kernel(const float* __restrict__ x, float* __restrict__ out,
                                float scale)
{
  const int64_t BE = (int64_t)Bz * Ez;
  const int64_t i = (int64_t)blockIdx.x * 256 + threadIdx.x;
  if (i >= BE) return;
  float s = 0.f;
  for (int ss = 0; ss < Ssz; ++ss) s += x[(int64_t)ss * BE + i];
  out[i] = s * scale;
}

// ---------------------------------------------------------------------------
extern "C" void kernel_launch(void* const* d_in, const int* in_sizes, int n_in,
                              void* d_out, int out_size, void* d_ws, size_t ws_size,
                              hipStream_t stream)
{
  (void)in_sizes; (void)n_in; (void)out_size; (void)ws_size;
  const int* premise    = (const int*)d_in[0];
  const int* hypothesis = (const int*)d_in[1];

  // ---- workspace layout: slots of AE floats (32 MiB) each ----
  const int64_t AE = (int64_t)Ssz * Bz * Ez;     // 8,388,608
  float* W = (float*)d_ws;
  auto slot = [&](int i) { return (void*)(W + (int64_t)i * AE); };

  float* e1  = (float*)slot(0);
  float* e2  = (float*)slot(1);
  float* i1  = (float*)slot(2);
  float* i2  = (float*)slot(3);
  h16*  hb16 = (h16*)slot(4);               // AE halves (first half of slot 4)
  h16*  scr16 = (h16*)slot(4) + AE;         // AE halves: V-transpose / cross-attn mem
  h16*  qv16 = (h16*)slot(5);
  h16*  kv16 = (h16*)slot(5) + AE;
  h16*  vv16 = (h16*)slot(6);
  h16*  ov16 = (h16*)slot(6) + AE;
  float* scores = (float*)slot(7);          // slots 7-10 (B*H*S*S fp32)
  h16*  ffh16 = (h16*)slot(7);              // slots 7-8 as f16 (FFN hidden / f1 out)
  h16*  probs16 = (h16*)slot(11);           // slots 11-12 (B*H*S*S f16)
  h16*  cat16 = (h16*)slot(13);             // slot 13 (M x 2E f16)
  float* f2out = (float*)slot(14);          // slot 14
  float* x1   = (float*)slot(15);
  float* x2   = x1 + (int64_t)Bz * Ez;
  h16*  xcat16 = (h16*)(x2 + (int64_t)Bz * Ez);
  h16*  f5o16  = xcat16 + (int64_t)Bz * 2 * Ez;
  h16*  wheap = (h16*)slot(16);             // slots 16-17: f16 weight heap (~46 MB)
  int64_t woff = 0;

  // ---- unpack params (setup_inputs dict order, depth-first) + f16 weights ----
  int pidx = 2;
  auto nf = [&]() { return (const float*)d_in[pidx++]; };
  auto cvtw = [&](int64_t n) -> const h16* {
    const float* src = nf();
    h16* dst = wheap + woff;
    woff += (n + 7) & ~7LL;
    cvt32to16_kernel<<<(unsigned)((n + 1023) / 1024), 256, 0, stream>>>(src, dst, n);
    return dst;
  };

  const float* embedding = nf();
  const float* proj_w    = nf();
  const float* proj_b    = nf();

  struct MHAp { const h16 *qw,*kw,*vw,*ow; const float *qb,*kb,*vb,*ob; };
  struct FFNp { const h16 *w1,*w2; const float *b1,*b2; };
  struct EncP { MHAp sa; FFNp ffn; const float *ln1g,*ln1b,*ln2g,*ln2b; };
  struct DecP { MHAp sa, ea; FFNp ffn;
                const float *ln1g,*ln1b,*ln2g,*ln2b,*ln3g,*ln3b; };

  auto rd_mha = [&](MHAp& m){
    m.qw = cvtw((int64_t)Ez * Ez); m.qb = nf();
    m.kw = cvtw((int64_t)Ez * Ez); m.kb = nf();
    m.vw = cvtw((int64_t)Ez * Ez); m.vb = nf();
    m.ow = cvtw((int64_t)Ez * Ez); m.ob = nf();
  };
  auto rd_ffn = [&](FFNp& f){
    f.w1 = cvtw((int64_t)FFNsz * Ez); f.b1 = nf();
    f.w2 = cvtw((int64_t)Ez * FFNsz); f.b2 = nf();
  };

  EncP enc[NLAY]; DecP dec[NLAY];
  for (int l = 0; l < NLAY; ++l) {
    rd_mha(enc[l].sa); rd_ffn(enc[l].ffn);
    enc[l].ln1g=nf(); enc[l].ln1b=nf(); enc[l].ln2g=nf(); enc[l].ln2b=nf();
  }
  for (int l = 0; l < NLAY; ++l) {
    rd_mha(dec[l].sa); rd_mha(dec[l].ea); rd_ffn(dec[l].ffn);
    dec[l].ln1g=nf(); dec[l].ln1b=nf(); dec[l].ln2g=nf(); dec[l].ln2b=nf();
    dec[l].ln3g=nf(); dec[l].ln3b=nf();
  }
  const h16* f1w = cvtw((int64_t)Ez * 2 * Ez);  const float* f1b = nf();
  const h16* f2w = cvtw((int64_t)Ez * Ez);      const float* f2b = nf();
  const h16* f3w = cvtw((int64_t)Ez * 2 * Ez);  const float* f3b = nf();
  const h16* f4w = cvtw((int64_t)Ez * Ez);      const float* f4b = nf();
  const h16* f5w = cvtw((int64_t)Ez * 2 * Ez);  const float* f5b = nf();
  const h16* f6w = cvtw((int64_t)UNITSZ * Ez);  const float* f6b = nf();

  const int M = Ssz * Bz;   // 16384 tokens
  const int64_t BE = (int64_t)Bz * Ez;

  auto gemmF = [&](const h16* A, int64_t lda, int64_t aBS,
                   const h16* Bw, int64_t ldb, int64_t bBS,
                   const float* bias, const float* resid,
                   void* C, int64_t ldc, int64_t cBS,
                   int m, int n, int k, int batch, float scale, int relu, int out16){
    dim3 g((unsigned)(n / 64), (unsigned)(m / 128), (unsigned)batch);
    if (out16)
      gemm_fast16_kernel<1><<<g, 256, 0, stream>>>(A, lda, aBS, Bw, ldb, bBS,
                                                   bias, resid, C, ldc, cBS, k, scale, relu);
    else
      gemm_fast16_kernel<0><<<g, 256, 0, stream>>>(A, lda, aBS, Bw, ldb, bBS,
                                                   bias, resid, C, ldc, cBS, k, scale, relu);
  };
  auto gemmG = [&](const h16* A, int64_t lda, const h16* Bw, int64_t ldb,
                   const float* bias, void* C, int64_t ldc,
                   int m, int n, int k, float scale, int relu, int out16){
    dim3 g((unsigned)((n + 15) / 16), (unsigned)((m + 127) / 128), 1);
    gemm_guarded16_kernel<<<g, 256, 0, stream>>>(A, lda, Bw, ldb, bias, C, ldc,
                                                 m, n, k, scale, relu, out16);
  };
  auto ln = [&](const float* x, const float* g, const float* b, h16* y){
    ln512_kernel<<<M, 256, 0, stream>>>(x, g, b, y);
  };
  auto mha = [&](const h16* xh, const h16* mem, const MHAp& pm, float* xio){
    gemmF(xh,  Ez, 0, pm.qw, Ez, 0, pm.qb, nullptr, qv16, Ez, 0, M, Ez, Ez, 1, 0.125f, 0, 1);
    gemmF(mem, Ez, 0, pm.kw, Ez, 0, pm.kb, nullptr, kv16, Ez, 0, M, Ez, Ez, 1, 1.0f,   0, 1);
    gemmF(mem, Ez, 0, pm.vw, Ez, 0, pm.vb, nullptr, vv16, Ez, 0, M, Ez, Ez, 1, 1.0f,   0, 1);
    // scores[z] = Q_z @ K_z^T ; head z base offset 64*z, row stride B*E
    gemmF(qv16, BE, HEADsz, kv16, BE, HEADsz, nullptr, nullptr,
          scores, Ssz, (int64_t)Ssz * Ssz, Ssz, Ssz, HEADsz, Bz * Hh, 1.0f, 0, 0);
    softmax256_kernel<<<Bz * Hh * Ssz, 256, 0, stream>>>(scores, probs16);
    // vt[z] = V_z^T  ->  weight-form for P@V
    transposeV_kernel<<<(unsigned)(((int64_t)Bz*Hh*HEADsz*Ssz + 255) / 256), 256, 0, stream>>>(
        vv16, scr16);
    gemmF(probs16, Ssz, (int64_t)Ssz * Ssz, scr16, Ssz, (int64_t)HEADsz * Ssz,
          nullptr, nullptr, ov16, BE, HEADsz, Ssz, HEADsz, Ssz, Bz * Hh, 1.0f, 0, 1);
    // output projection + residual add into xio (fp32)
    gemmF(ov16, Ez, 0, pm.ow, Ez, 0, pm.ob, xio, xio, Ez, 0, M, Ez, Ez, 1, 1.0f, 0, 0);
  };
  auto ffn = [&](const h16* xh, const FFNp& pf, float* xio){
    gemmF(xh,   Ez,    0, pf.w1, Ez,    0, pf.b1, nullptr, ffh16, FFNsz, 0,
          M, FFNsz, Ez, 1, 1.0f, 1, 1);
    gemmF(ffh16, FFNsz, 0, pf.w2, FFNsz, 0, pf.b2, xio,    xio,   Ez,    0,
          M, Ez, FFNsz, 1, 1.0f, 0, 0);
  };
  auto enc_layer = [&](float* x, const EncP& pe){
    ln(x, pe.ln1g, pe.ln1b, hb16); mha(hb16, hb16, pe.sa, x);
    ln(x, pe.ln2g, pe.ln2b, hb16); ffn(hb16, pe.ffn, x);
  };
  auto dec_layer = [&](float* x, const float* mem, const DecP& pd){
    ln(x, pd.ln1g, pd.ln1b, hb16); mha(hb16, hb16, pd.sa, x);
    ln(x, pd.ln2g, pd.ln2b, hb16);
    // cross-attn k/v come from raw (non-LN) memory stream: f16 mirror into scr16
    cvt32to16_kernel<<<(unsigned)((AE + 1023) / 1024), 256, 0, stream>>>(mem, scr16, AE);
    mha(hb16, scr16, pd.ea, x);   // scr16 re-used as vt inside mha AFTER k/v gemms
    ln(x, pd.ln3g, pd.ln3b, hb16); ffn(hb16, pd.ffn, x);
  };

  // ---- forward pass ----
  embed_kernel<<<Bz * Ssz, 256, 0, stream>>>(premise,    embedding, proj_w, proj_b, e1);
  embed_kernel<<<Bz * Ssz, 256, 0, stream>>>(hypothesis, embedding, proj_w, proj_b, e2);

  for (int l = 0; l < NLAY; ++l) { enc_layer(e1, enc[l]); enc_layer(e2, enc[l]); }

  (void)hipMemcpyAsync(i1, e1, (size_t)AE * sizeof(float), hipMemcpyDeviceToDevice, stream);
  (void)hipMemcpyAsync(i2, e2, (size_t)AE * sizeof(float), hipMemcpyDeviceToDevice, stream);

  for (int l = 0; l < NLAY; ++l) { dec_layer(i1, i2, dec[l]); dec_layer(i2, i1, dec[l]); }

  auto branch = [&](const float* encx, const float* interx,
                    const h16* w1, const float* b1,
                    const h16* w2, const float* b2, float* xout){
    const int64_t total = (int64_t)M * 2 * Ez;
    concat2_kernel<<<(unsigned)((total + 255) / 256), 256, 0, stream>>>(encx, interx, cat16, M);
    gemmF(cat16, 2 * Ez, 0, w1, 2 * Ez, 0, b1, nullptr, ffh16, Ez, 0,
          M, Ez, 2 * Ez, 1, 1.0f, 1, 1);
    gemmF(ffh16, Ez,     0, w2, Ez,     0, b2, nullptr, f2out, Ez, 0,
          M, Ez, Ez, 1, 1.0f, 0, 0);
    sumscale_kernel<<<(unsigned)((BE + 255) / 256), 256, 0, stream>>>(
        f2out, xout, 1.0f / ((float)Ssz * (float)Ssz));
  };
  branch(e1, i1, f1w, f1b, f2w, f2b, x1);
  branch(e2, i2, f3w, f3b, f4w, f4b, x2);

  {
    const int64_t total = (int64_t)Bz * 2 * Ez;
    concat2_kernel<<<(unsigned)((total + 255) / 256), 256, 0, stream>>>(x1, x2, xcat16, Bz);
  }
  gemmG(xcat16, 2 * Ez, f5w, 2 * Ez, f5b, f5o16, Ez, Bz, Ez, 2 * Ez, 1.0f, 1, 1);
  gemmG(f5o16, Ez, f6w, Ez, f6b, (float*)d_out, UNITSZ, Bz, UNITSZ, Ez, 1.0f, 0, 0);
}
